// OcclusionLayer_66597762892542
// MI455X (gfx1250) — matrix operand order repaired and checked by reading, not compiled, not run on hardware
//
#include <hip/hip_runtime.h>

// ---------------------------------------------------------------------------
// 4-layer MLP, batch 256:
//   h1  = relu(x  @ W1.T + b1)   [256, 515]      (tiny, VALU kernel)
//   h2b = relu(h1 @ W2.T + b2)   [256, 288] bf16 (zero-padded K for WMMA)
//   h3  = relu(h2 @ W3.T + b3)   [256, 65536] bf16 (WMMA)
//   out = relu(h3 @ W4.T + b4)   [256, 2048] f32  (WMMA)
// W4 = 512MB fp32 streams from HBM once (~22us @ 23.3TB/s floor); h3 kept
// as bf16 (32MB) in workspace so it stays resident in the 192MB L2.
// ---------------------------------------------------------------------------

typedef __attribute__((ext_vector_type(16))) __bf16 v16bf;
typedef __attribute__((ext_vector_type(8)))  __bf16 bf16x8;
typedef __attribute__((ext_vector_type(8)))  float  v8f;
typedef __attribute__((ext_vector_type(4)))  float  f32x4;

#define BATCH 256
#define S0 7
#define S1 515
#define S2 260
#define S2P 288   // S2 rounded up to multiple of 32 (WMMA K step)
#define S3 65536
#define SOUT 2048

// ---------------------------------------------------------------------------
// Small layers.
// ---------------------------------------------------------------------------
__global__ __launch_bounds__(256) void fc1_kernel(
    const float* __restrict__ x, const float* __restrict__ W1,
    const float* __restrict__ b1, float* __restrict__ h1) {
  int idx = blockIdx.x * blockDim.x + threadIdx.x;
  if (idx >= BATCH * S1) return;
  int b = idx / S1, j = idx % S1;
  float s = b1[j];
  const float* xr = x + b * S0;
  const float* wr = W1 + j * S0;
#pragma unroll
  for (int i = 0; i < S0; ++i) s = fmaf(xr[i], wr[i], s);
  h1[idx] = s > 0.0f ? s : 0.0f;
}

// Writes h2 as zero-padded bf16 [BATCH, S2P]: cols >= S2 are 0.
__global__ __launch_bounds__(256) void fc2_kernel(
    const float* __restrict__ h1, const float* __restrict__ W2,
    const float* __restrict__ b2, __bf16* __restrict__ h2b) {
  int idx = blockIdx.x * blockDim.x + threadIdx.x;
  if (idx >= BATCH * S2P) return;
  int b = idx / S2P, n = idx % S2P;
  float s = 0.0f;
  if (n < S2) {
    const float* hr = h1 + b * S1;
    const float* wr = W2 + n * S1;
    s = b2[n];
    for (int k = 0; k < S1; ++k) s = fmaf(hr[k], wr[k], s);
    s = s > 0.0f ? s : 0.0f;
  }
  h2b[idx] = (__bf16)s;
}

// ---------------------------------------------------------------------------
// Per-lane fragment loaders (contiguous k = 16*(lane/16) + s, same order for A and B).
// ---------------------------------------------------------------------------

// B fragment: 16 contiguous fp32 -> bf16 (K run 16*(lane/16) .. +15).
__device__ __forceinline__ v16bf load_b_fast(const float* __restrict__ row,
                                             int off) {
  const f32x4* p = (const f32x4*)(row + off);
  f32x4 a = p[0], b = p[1], c = p[2], d = p[3];
  v16bf v;
#pragma unroll
  for (int e = 0; e < 4; ++e) {
    v[e]      = (__bf16)a[e];
    v[4 + e]  = (__bf16)b[e];
    v[8 + e]  = (__bf16)c[e];
    v[12 + e] = (__bf16)d[e];
  }
  return v;
}

// B fragment, K tail: branchless guard (clamped index + select-to-zero) so
// loads stay unconditional -> no EXEC divergence in a WMMA kernel.
__device__ __forceinline__ v16bf load_b_tail(const float* __restrict__ row,
                                             int base, int K) {
  v16bf v;
#pragma unroll
  for (int e = 0; e < 16; ++e) {
    int k  = base + e;
    int kc = (k < K) ? k : 0;   // always in-bounds
    float f = row[kc];
    v[e] = (__bf16)((k < K) ? f : 0.0f);
  }
  return v;
}

// A fragment from bf16 memory: two 16B loads (runs {b1..+7},{b1+8..+7}).
__device__ __forceinline__ v16bf load_a_bf16(const __bf16* __restrict__ row,
                                             int b1) {
  bf16x8 lo = *(const bf16x8*)(row + b1);
  bf16x8 hi = *(const bf16x8*)(row + b1 + 8);
  return __builtin_shufflevector(lo, hi, 0, 1, 2, 3, 4, 5, 6, 7,
                                 8, 9, 10, 11, 12, 13, 14, 15);
}

// ---------------------------------------------------------------------------
// WMMA GEMM: out[256, N] = relu(A[256, Kpad] @ B[N, K]^T + bias[N])
// A is bf16 with lda >= roundup32(K) (zero-padded). B is fp32, converted
// to bf16 in registers (W4 must stream from HBM exactly once).
// block = 128 threads (4 waves); block owns one 16-wide N tile,
// wave w owns rows [64w, 64w+64) as 4 accumulator tiles.
// ---------------------------------------------------------------------------
template <bool OUT_BF16>
__global__ __launch_bounds__(128, 2) void gemm_relu_wmma(
    const __bf16* __restrict__ A, long lda, const float* __restrict__ B,
    long ldb, const float* __restrict__ bias, void* __restrict__ outv,
    long ldout, int K) {
  const int lane   = threadIdx.x & 31;
  const int wave   = threadIdx.x >> 5;
  const int laneLo = lane & 15;
  const int laneHi = lane >> 4;
  const long nBase = (long)blockIdx.x * 16;
  const int mWave  = wave * 64;

  const float* Brow = B + (nBase + laneLo) * ldb;
  const __bf16* Arow[4];
#pragma unroll
  for (int t = 0; t < 4; ++t)
    Arow[t] = A + (long)(mWave + t * 16 + laneLo) * lda;

  const int aOff = laneHi * 16;  // A K-run start within 32-wide k-block
  const int bOff = laneHi * 16;  // B K-run start within 32-wide k-block

  v8f z = {0.f, 0.f, 0.f, 0.f, 0.f, 0.f, 0.f, 0.f};
  v8f acc[4];
#pragma unroll
  for (int t = 0; t < 4; ++t) acc[t] = z;

  const int kFull = K & ~31;
  for (int kb = 0; kb < kFull; kb += 32) {
    v16bf bmat = load_b_fast(Brow, kb + bOff);
#pragma unroll
    for (int t = 0; t < 4; ++t) {
      v16bf amat = load_a_bf16(Arow[t], kb + aOff);
      acc[t] = __builtin_amdgcn_wmma_f32_16x16x32_bf16(
          false, amat, false, bmat, (short)0, acc[t], false, false);
    }
  }
  if (K & 31) {  // uniform branch; A is zero-padded so only B needs the guard
    v16bf bmat = load_b_tail(Brow, kFull + bOff, K);
#pragma unroll
    for (int t = 0; t < 4; ++t) {
      v16bf amat = load_a_bf16(Arow[t], kFull + aOff);
      acc[t] = __builtin_amdgcn_wmma_f32_16x16x32_bf16(
          false, amat, false, bmat, (short)0, acc[t], false, false);
    }
  }

  const float bv = bias[nBase + laneLo];
#pragma unroll
  for (int t = 0; t < 4; ++t) {
#pragma unroll
    for (int r = 0; r < 8; ++r) {
      const long m = mWave + t * 16 + r + 8 * laneHi;
      float v = acc[t][r] + bv;
      v = v > 0.0f ? v : 0.0f;
      if constexpr (OUT_BF16)
        ((__bf16*)outv)[m * ldout + nBase + laneLo] = (__bf16)v;
      else
        ((float*)outv)[m * ldout + nBase + laneLo] = v;
    }
  }
}

// ---------------------------------------------------------------------------
// Launch
// ---------------------------------------------------------------------------
extern "C" void kernel_launch(void* const* d_in, const int* in_sizes, int n_in,
                              void* d_out, int out_size, void* d_ws,
                              size_t ws_size, hipStream_t stream) {
  const float* x  = (const float*)d_in[0];
  const float* W1 = (const float*)d_in[1];
  const float* b1 = (const float*)d_in[2];
  const float* W2 = (const float*)d_in[3];
  const float* b2 = (const float*)d_in[4];
  const float* W3 = (const float*)d_in[5];
  const float* b3 = (const float*)d_in[6];
  const float* W4 = (const float*)d_in[7];
  const float* b4 = (const float*)d_in[8];

  // Workspace: h1 (fp32) | h2b (bf16, zero-padded K) | h3 (bf16, 32MB)
  float*  h1  = (float*)d_ws;
  __bf16* h2b = (__bf16*)(h1 + (size_t)BATCH * S1);
  __bf16* h3  = h2b + (size_t)BATCH * S2P;

  fc1_kernel<<<(BATCH * S1 + 255) / 256, 256, 0, stream>>>(x, W1, b1, h1);
  fc2_kernel<<<(BATCH * S2P + 255) / 256, 256, 0, stream>>>(h1, W2, b2, h2b);

  // L3: A=h2b bf16 [256,288], K=260 (8 fast K-steps + 1 guarded tail)
  gemm_relu_wmma<true><<<S3 / 16, 128, 0, stream>>>(
      h2b, (long)S2P, W3, (long)S2, b3, (void*)h3, (long)S3, S2);

  // L4: A=h3 bf16 [256,65536], W4 fp32 converted in-register, out fp32
  gemm_relu_wmma<false><<<SOUT / 16, 128, 0, stream>>>(
      h3, (long)S3, W4, (long)S3, b4, d_out, (long)SOUT, S3);
}
